// AtlasMemoryLayer_65377992180186
// MI455X (gfx1250) — matrix operand order repaired and checked by reading, not compiled, not run on hardware
//
#include <hip/hip_runtime.h>
#include <hip/hip_bf16.h>
#include <math.h>

// ---------------- problem constants (from reference) ----------------
#define HN    8     // heads
#define DD    64    // head dim
#define CC    512   // model dim
#define CSZ   64    // chunk size
#define NSTEP 5     // Newton-Schulz steps
#define OWIN  8     // omega window
#define PDEG  5     // poly degree
#define KCV   4     // conv kernel
#define BB    2
#define TT    1024
#define NCHUNK (TT / CSZ)
#define MAXLR 0.1f

#define LDP 68      // padded f32 LDS leading dim (64+4): conflict-free frag loads
#define LDW 36      // packed-bf16 LDS leading dim in dwords (32+4): 16B aligned, conflict-free

typedef __attribute__((ext_vector_type(16))) __bf16 v16bf;
typedef __attribute__((ext_vector_type(8)))  float  v8f;
typedef __attribute__((ext_vector_type(4)))  unsigned int u32x4;
typedef __attribute__((ext_vector_type(8)))  int    i32x8;
typedef __attribute__((ext_vector_type(4)))  int    i32x4;

// =================== TDM (Tensor Data Mover) support ===================
#if defined(__HIP_DEVICE_COMPILE__) && __has_builtin(__builtin_amdgcn_tensor_load_to_lds)
#define HAVE_TDM 1
#endif

#ifdef HAVE_TDM
// Async-load a 64x64 f32 tile (global rows strided by row_stride_elems) into LDS
// rows of LDP floats (pad_amount=4 dwords after every pad_interval=64 dwords).
// D# per CDNA5 ISA. Issue once per workgroup (wave 0);
// TDM ignores EXEC. Track with TENSORcnt.
__device__ __forceinline__ void tdm_load_tile64(const float* gsrc, unsigned lds_off,
                                                unsigned row_stride_elems) {
  unsigned long long ga = (unsigned long long)(const void*)gsrc;
  u32x4 g0;
  g0[0] = 1u;                                              // count=1, user descriptor
  g0[1] = lds_off;                                         // LDS byte address
  g0[2] = (unsigned)(ga & 0xFFFFFFFFull);                  // global addr lo
  g0[3] = (unsigned)((ga >> 32) & 0x1FFFFFFull) | (2u << 30); // addr hi | type=2
  i32x8 g1;
  g1[0] = (int)((2u << 16)      // data_size = 4B
              | (1u << 20)      // pad_enable
              | (5u << 22)      // pad_interval: 64 dwords
              | (3u << 25));    // pad_amount: 4 dwords
  g1[1] = (int)(64u << 16);     // tensor_dim0 = 64 (bits 79:48)
  g1[2] = (int)(64u << 16);     // tensor_dim1 = 64 (bits 111:80)
  g1[3] = (int)(64u << 16);     // tile_dim0 = 64  (bits 127:112)
  g1[4] = 64;                   // tile_dim1 = 64, tile_dim2 = 0
  g1[5] = (int)row_stride_elems;// tensor_dim0_stride (low 32)
  g1[6] = 0;
  g1[7] = 0;
  i32x4 z = {0, 0, 0, 0};
#if defined(__clang_major__) && __clang_major__ >= 23
  i32x8 z8 = {0, 0, 0, 0, 0, 0, 0, 0};
  __builtin_amdgcn_tensor_load_to_lds(g0, g1, z, z, z8, 0);
#else
  __builtin_amdgcn_tensor_load_to_lds(g0, g1, z, z, 0);
#endif
}
#define LDS_BYTE_OFF(p) ((unsigned)(unsigned long long)(const void*)(p))
#endif

// ---------------- WMMA bf16 fragment helpers (CDNA5 ISA layouts) ----
// A: 16x32 bf16. lane&15 = M row; lane>>4 selects K-halves {0-7,16-23} vs {8-15,24-31}.
__device__ __forceinline__ v16bf frag_a(const float* A, int lda, int m0, int k0, int lane) {
  const int m  = m0 + (lane & 15);
  const int kh = k0 + ((lane >> 4) << 3);
  const float* row = A + (size_t)m * lda;
  v16bf f;
#pragma unroll
  for (int v = 0; v < 8; ++v) {
    const int kb = kh + ((v & 3) << 1) + ((v >> 2) << 4);
    f[2 * v]     = (__bf16)row[kb];
    f[2 * v + 1] = (__bf16)row[kb + 1];
  }
  return f;
}

// B: 32x16 bf16. lane&15 = N col; lane>>4 selects K-halves {0-7,16-23} vs {8-15,24-31}.
__device__ __forceinline__ v16bf frag_b(const float* B, int ldb, int k0, int n0,
                                        int lane, bool trans) {
  const int n  = n0 + (lane & 15);
  const int kh = k0 + ((lane >> 4) << 3);
  v16bf f;
#pragma unroll
  for (int v = 0; v < 8; ++v) {
    const int kb = kh + ((v & 3) << 1) + ((v >> 2) << 4);
    float e0, e1;
    if (trans) { e0 = B[(size_t)n * ldb + kb];  e1 = B[(size_t)n * ldb + kb + 1]; }
    else       { e0 = B[(size_t)kb * ldb + n];  e1 = B[(size_t)(kb + 1) * ldb + n]; }
    f[2 * v]     = (__bf16)e0;
    f[2 * v + 1] = (__bf16)e1;
  }
  return f;
}

// --- packed-bf16 fragment loaders: P rows hold K-pairs as dwords (2 bf16/dword) ---
// A frag from row-major packed: dword index = k/2 within row m.
__device__ __forceinline__ v16bf frag_a_pk(const unsigned* P, int m0, int k0, int lane) {
  const int m   = m0 + (lane & 15);
  const int d0  = (k0 >> 1) + ((lane >> 4) << 2);
  const unsigned* row = P + (size_t)m * LDW + d0;
  union { unsigned u[8]; v16bf f; } r;
#pragma unroll
  for (int v = 0; v < 4; ++v) r.u[v] = row[v];         // K pairs kh+0..7
#pragma unroll
  for (int v = 0; v < 4; ++v) r.u[4 + v] = row[8 + v]; // K pairs kh+16..23
  return r.f;
}
// B frag: P row n holds B(k,n) pairs along k (i.e. P is B^T packed row-major).
__device__ __forceinline__ v16bf frag_b_pk(const unsigned* P, int k0, int n0, int lane) {
  const int n  = n0 + (lane & 15);
  const int d0 = (k0 >> 1) + ((lane >> 4) << 2);
  const unsigned* row = P + (size_t)n * LDW + d0;
  union { unsigned u[8]; v16bf f; } r;
#pragma unroll
  for (int v = 0; v < 8; ++v) r.u[v] = row[(v & 3) + ((v >> 2) << 3)];
  return r.f;
}

union bfpair { unsigned u; __bf16 h[2]; };
__device__ __forceinline__ unsigned pack2(float a, float b) {
  bfpair p; p.h[0] = (__bf16)a; p.h[1] = (__bf16)b; return p.u;
}
// row-major pack: P[r][d] = (F[r][2d], F[r][2d+1]); 256 threads cover 64x32 dwords
__device__ __forceinline__ void pack_rm(const float* F, int ldf, unsigned* P) {
  const int r = threadIdx.x >> 2, d0 = (threadIdx.x & 3) << 3;
  const float* row = F + (size_t)r * ldf + (d0 << 1);
  unsigned* prow = P + (size_t)r * LDW + d0;
#pragma unroll
  for (int j = 0; j < 8; ++j) prow[j] = pack2(row[2 * j], row[2 * j + 1]);
}
// transposed pack: P[c][d] = (F[2d][c], F[2d+1][c])
__device__ __forceinline__ void pack_tr(const float* F, int ldf, unsigned* P) {
  const int c = threadIdx.x >> 2, d0 = (threadIdx.x & 3) << 3;
  unsigned* prow = P + (size_t)c * LDW + d0;
#pragma unroll
  for (int j = 0; j < 8; ++j)
    prow[j] = pack2(F[(size_t)(2 * (d0 + j)) * ldf + c],
                    F[(size_t)(2 * (d0 + j) + 1) * ldf + c]);
}

// Block-level 64x64x64 GEMM from f32 LDS/global operands (8 waves, 2 tiles/wave).
__device__ __forceinline__ void block_gemm64(const float* A, int lda,
                                             const float* B, int ldb, bool trans,
                                             float* C, int ldc) {
  const int wave = threadIdx.x >> 5;
  const int lane = threadIdx.x & 31;
#pragma unroll
  for (int s = 0; s < 2; ++s) {
    const int tt = wave * 2 + s;
    const int tm = (tt >> 2) << 4;
    const int tn = (tt & 3) << 4;
    v8f acc = {};
#pragma unroll
    for (int kk = 0; kk < 64; kk += 32) {
      v16bf a = frag_a(A, lda, tm, kk, lane);
      v16bf b = frag_b(B, ldb, kk, tn, lane, trans);
      acc = __builtin_amdgcn_wmma_f32_16x16x32_bf16(false, a, false, b,
                                                    (short)0, acc, false, false);
    }
    const int n  = tn + (lane & 15);
    const int mh = tm + ((lane >> 4) << 3);
#pragma unroll
    for (int r = 0; r < 8; ++r) C[(size_t)(mh + r) * ldc + n] = acc[r];
  }
}

// GEMM from packed-bf16 operands: C = A * B (Apk row-major packed, Bpk = B^T packed)
__device__ __forceinline__ void block_gemm64_pk(const unsigned* Apk, const unsigned* Bpk,
                                                float* C, int ldc) {
  const int wave = threadIdx.x >> 5, lane = threadIdx.x & 31;
#pragma unroll
  for (int s = 0; s < 2; ++s) {
    const int tt = wave * 2 + s;
    const int tm = (tt >> 2) << 4, tn = (tt & 3) << 4;
    v8f acc = {};
#pragma unroll
    for (int kk = 0; kk < 64; kk += 32) {
      v16bf a = frag_a_pk(Apk, tm, kk, lane);
      v16bf b = frag_b_pk(Bpk, kk, tn, lane);
      acc = __builtin_amdgcn_wmma_f32_16x16x32_bf16(false, a, false, b,
                                                    (short)0, acc, false, false);
    }
    const int n = tn + (lane & 15), mh = tm + ((lane >> 4) << 3);
#pragma unroll
    for (int r = 0; r < 8; ++r) C[(size_t)(mh + r) * ldc + n] = acc[r];
  }
}

// Same, but epilogue fuses the quintic combine: X = a*X + b*Y1 + c*acc (acc = A*Y1)
__device__ __forceinline__ void block_gemm64_pk_polar(const unsigned* Apk, const unsigned* Bpk,
                                                      const float* Y1, int ldy,
                                                      float* X, int ldx) {
  const float aC = 3.4445f, bC = -4.7750f, cC = 2.0315f;
  const int wave = threadIdx.x >> 5, lane = threadIdx.x & 31;
#pragma unroll
  for (int s = 0; s < 2; ++s) {
    const int tt = wave * 2 + s;
    const int tm = (tt >> 2) << 4, tn = (tt & 3) << 4;
    v8f acc = {};
#pragma unroll
    for (int kk = 0; kk < 64; kk += 32) {
      v16bf a = frag_a_pk(Apk, tm, kk, lane);
      v16bf b = frag_b_pk(Bpk, kk, tn, lane);
      acc = __builtin_amdgcn_wmma_f32_16x16x32_bf16(false, a, false, b,
                                                    (short)0, acc, false, false);
    }
    const int n = tn + (lane & 15), mh = tm + ((lane >> 4) << 3);
#pragma unroll
    for (int r = 0; r < 8; ++r) {
      const int m = mh + r;
      X[(size_t)m * ldx + n] = aC * X[(size_t)m * ldx + n]
                             + bC * Y1[(size_t)m * ldy + n] + cC * acc[r];
    }
  }
}

// ---------------- kernels ----------------

// Out[r][n] = sum_c X[r][c] * W[n][c]   (rows = gridDim.x*64, N = gridDim.y*64)
__global__ __launch_bounds__(256) void k_proj(const float* __restrict__ X,
                                              const float* __restrict__ W,
                                              float* __restrict__ Out, int K) {
  const int r0 = blockIdx.x << 6, n0 = blockIdx.y << 6;
  const int wave = threadIdx.x >> 5, lane = threadIdx.x & 31;
  const int N = gridDim.y << 6;
  __shared__ float As[64][LDP];
  v8f acc0 = {}; v8f acc1 = {};
  const int tt0 = wave * 2, tt1 = wave * 2 + 1;
  const int tm0 = (tt0 >> 2) << 4, tn0 = (tt0 & 3) << 4;
  const int tm1 = (tt1 >> 2) << 4, tn1 = (tt1 & 3) << 4;
  for (int k0 = 0; k0 < K; k0 += 64) {
    __syncthreads();
    for (int i = threadIdx.x; i < 64 * 64; i += 256) {
      const int rr = i >> 6, cc = i & 63;
      As[rr][cc] = X[(size_t)(r0 + rr) * K + k0 + cc];
    }
    __syncthreads();
#pragma unroll
    for (int kk = 0; kk < 64; kk += 32) {
      v16bf a0 = frag_a(&As[0][0], LDP, tm0, kk, lane);
      v16bf b0 = frag_b(W, K, k0 + kk, n0 + tn0, lane, true);
      acc0 = __builtin_amdgcn_wmma_f32_16x16x32_bf16(false, a0, false, b0,
                                                     (short)0, acc0, false, false);
      v16bf a1 = frag_a(&As[0][0], LDP, tm1, kk, lane);
      v16bf b1 = frag_b(W, K, k0 + kk, n0 + tn1, lane, true);
      acc1 = __builtin_amdgcn_wmma_f32_16x16x32_bf16(false, a1, false, b1,
                                                     (short)0, acc1, false, false);
    }
  }
  const int n = lane & 15, mh = (lane >> 4) << 3;
#pragma unroll
  for (int r = 0; r < 8; ++r) {
    Out[(size_t)(r0 + tm0 + mh + r) * N + n0 + tn0 + n] = acc0[r];
    Out[(size_t)(r0 + tm1 + mh + r) * N + n0 + tn1 + n] = acc1[r];
  }
}

// causal depthwise conv (K=4) + bias + optional phi polynomial
__global__ __launch_bounds__(256) void k_conv(const float* __restrict__ pre,
                                              const float* __restrict__ w,
                                              const float* __restrict__ bias,
                                              const float* __restrict__ poly,
                                              float* __restrict__ out, int apply_phi) {
  const int idx = blockIdx.x * 256 + threadIdx.x;   // B*T*512
  const int din = idx & 511;
  const int row = idx >> 9;          // b*T + t
  const int t   = row & (TT - 1);
  const float w0 = w[din * KCV + 0], w1 = w[din * KCV + 1];
  const float w2 = w[din * KCV + 2], w3 = w[din * KCV + 3];
  float xm3 = (t >= 3) ? pre[(size_t)(row - 3) * CC + din] : 0.f;
  float xm2 = (t >= 2) ? pre[(size_t)(row - 2) * CC + din] : 0.f;
  float xm1 = (t >= 1) ? pre[(size_t)(row - 1) * CC + din] : 0.f;
  float x0  = pre[(size_t)row * CC + din];
  float acc = bias[din] + w0 * xm3 + w1 * xm2 + w2 * xm1 + w3 * x0;
  if (apply_phi) {
    float z = acc, zp = z, r = poly[0] * z;
#pragma unroll
    for (int i = 1; i < PDEG; ++i) { zp *= z; r += poly[i] * zp; }
    acc = r;
  }
  out[idx] = acc;
}

// all five gates: alpha, eta (xMAX_LR), theta, gamma, rg
__global__ __launch_bounds__(64) void k_gates(const float* __restrict__ x,
    const float* __restrict__ gaw, const float* __restrict__ gab,
    const float* __restrict__ gew, const float* __restrict__ geb,
    const float* __restrict__ gtw, const float* __restrict__ gtb,
    const float* __restrict__ ggw, const float* __restrict__ ggb,
    const float* __restrict__ rgw,
    float* __restrict__ oa, float* __restrict__ oe, float* __restrict__ ot,
    float* __restrict__ og, float* __restrict__ orr) {
  __shared__ float xs[CC];
  const int bt = blockIdx.x;
  for (int i = threadIdx.x; i < CC; i += 64) xs[i] = x[(size_t)bt * CC + i];
  __syncthreads();
  const int tid = threadIdx.x;
  if (tid < 40) {
    const int g = tid >> 3, h = tid & 7;
    const float* wrow; const float* bb; float* outp;
    if      (g == 0) { wrow = gaw; bb = gab;     outp = oa;  }
    else if (g == 1) { wrow = gew; bb = geb;     outp = oe;  }
    else if (g == 2) { wrow = gtw; bb = gtb;     outp = ot;  }
    else if (g == 3) { wrow = ggw; bb = ggb;     outp = og;  }
    else             { wrow = rgw; bb = nullptr; outp = orr; }
    const float* wr = wrow + (size_t)h * CC;
    float d = 0.f;
    for (int c = 0; c < CC; ++c) d += xs[c] * wr[c];
    if (bb) d += bb[h];
    float s = 1.f / (1.f + __expf(-d));
    if (g == 1) s *= MAXLR;
    outp[(size_t)bt * HN + h] = s;
  }
}

// init M carry from W1_init (broadcast over batch) and zero S carry
__global__ __launch_bounds__(256) void k_init(const float* __restrict__ W1,
                                              float* __restrict__ Mst,
                                              float* __restrict__ Sst) {
  const int i = blockIdx.x * 256 + threadIdx.x;    // B*H*4096 = 65536
  const int h = (i >> 12) & 7;
  Mst[i] = W1[h * 4096 + (i & 4095)];
  Sst[i] = 0.f;
}

// per chunk: pred = Kc * M^T (WMMA), err = pred - v, momentum scan -> chunk_S
__global__ __launch_bounds__(256) void k_err_scan(
    const float* __restrict__ kb, const float* __restrict__ vb,
    const float* __restrict__ Mst, float* __restrict__ Sst,
    float* __restrict__ cS,
    const float* __restrict__ gth, const float* __restrict__ get,
    const float* __restrict__ ggm, int chunk) {
  __shared__ float Ks[64][LDP], Ms[64][LDP], Es[64][LDP];
  __shared__ float th[CSZ], et[CSZ], gm[CSZ];
  const int bh = blockIdx.x, b = bh >> 3, h = bh & 7;
  const int t0 = chunk * CSZ, tid = threadIdx.x;
#ifdef HAVE_TDM
  if (tid == 0) {
    tdm_load_tile64(kb + ((size_t)(b * TT + t0) * HN + h) * DD,
                    LDS_BYTE_OFF(&Ks[0][0]), HN * DD);
    tdm_load_tile64(Mst + (size_t)bh * 4096, LDS_BYTE_OFF(&Ms[0][0]), DD);
  }
  __builtin_amdgcn_s_wait_tensorcnt(0);
#else
  for (int i = tid; i < 4096; i += 256) {
    const int t = i >> 6, d = i & 63;
    Ks[t][d] = kb[((size_t)(b * TT + t0 + t) * HN + h) * DD + d];
    Ms[t][d] = Mst[(size_t)bh * 4096 + i];   // row index = v
  }
#endif
  if (tid < CSZ) {
    const size_t gi = (size_t)(b * TT + t0 + tid) * HN + h;
    th[tid] = gth[gi]; et[tid] = get[gi]; gm[tid] = ggm[gi];
  }
  __syncthreads();
  // pred[t][v] = sum_k Kc[t][k] * M[v][k]  -> B(k,v)=M[v][k]: trans=true
  block_gemm64(&Ks[0][0], LDP, &Ms[0][0], LDP, true, &Es[0][0], LDP);
  __syncthreads();
  for (int i = tid; i < 4096; i += 256) {
    const int t = i >> 6, d = i & 63;
    Es[t][d] -= vb[((size_t)(b * TT + t0 + t) * HN + h) * DD + d];
  }
  __syncthreads();
  // momentum scan with rank-8 omega window; each thread owns 16 S elements
  float S[16];
  const size_t sb = (size_t)bh * 4096 + (size_t)tid * 16;
#pragma unroll
  for (int j = 0; j < 16; ++j) S[j] = Sst[sb + j];
  const int v = tid >> 2, seg = (tid & 3) << 4;
  float* ob = cS + (size_t)bh * CSZ * 4096;
  const float sc = 2.0f / (float)DD;
  for (int t = 0; t < CSZ; ++t) {
    const float tht = th[t], ng = -et[t] * sc;
    int jlo = t - (OWIN - 1); if (jlo < 0) jlo = 0;
    float* orow = ob + (size_t)t * 4096 + (size_t)tid * 16;
#pragma unroll
    for (int j = 0; j < 16; ++j) {
      float acc = 0.f;
      for (int jj = jlo; jj <= t; ++jj) acc += gm[jj] * Es[jj][v] * Ks[jj][seg + j];
      const float s = tht * S[j] + ng * acc;
      S[j] = s;
      orow[j] = s;
    }
  }
#pragma unroll
  for (int j = 0; j < 16; ++j) Sst[sb + j] = S[j];
}

// Newton-Schulz quintic orthogonalization of one 64x64 matrix per block (in place).
// X' = aX + b*(A X) + c*(A*(A X)), A = X X^T; operands staged as packed bf16 in LDS.
__global__ __launch_bounds__(256) void k_ns(float* __restrict__ cS) {
  __shared__ float Xs[64][LDP];          // X (f32, persistent)
  __shared__ float Cf[64][LDP];          // GEMM f32 output (A, then Y1)
  __shared__ unsigned XpkR[64 * LDW];    // X packed row-major (also reused for Y1^T)
  __shared__ unsigned XpkT[64 * LDW];    // X packed transposed
  __shared__ unsigned Apk[64 * LDW];     // A packed row-major (A symmetric)
  __shared__ float red[9];
  float* mat = cS + (size_t)blockIdx.x * 4096;
  const int tid = threadIdx.x, wave = tid >> 5, lane = tid & 31;
  float ss = 0.f;
#ifdef HAVE_TDM
  if (tid == 0) tdm_load_tile64(mat, LDS_BYTE_OFF(&Xs[0][0]), DD);
  __builtin_amdgcn_s_wait_tensorcnt(0);
  __syncthreads();
  for (int i = tid; i < 4096; i += 256) {
    const float x = Xs[i >> 6][i & 63];
    ss += x * x;
  }
#else
  for (int i = tid; i < 4096; i += 256) {
    const float x = mat[i];
    Xs[i >> 6][i & 63] = x;
    ss += x * x;
  }
#endif
#pragma unroll
  for (int o = 16; o > 0; o >>= 1) ss += __shfl_down(ss, o, 32);
  if (lane == 0) red[wave] = ss;
  __syncthreads();
  if (tid == 0) {
    float tot = 0.f;
#pragma unroll
    for (int i = 0; i < 8; ++i) tot += red[i];
    red[8] = 1.0f / (sqrtf(tot) * 1.01f + 1e-7f);
  }
  __syncthreads();
  const float inv = red[8];
  for (int i = tid; i < 4096; i += 256) Xs[i >> 6][i & 63] *= inv;
  __syncthreads();
  pack_rm(&Xs[0][0], LDP, XpkR);
  pack_tr(&Xs[0][0], LDP, XpkT);
  __syncthreads();
  for (int it = 0; it < NSTEP; ++it) {
    // A = X X^T : B(k,n) = X[n][k] -> B^T packed == XpkR
    block_gemm64_pk(XpkR, XpkR, &Cf[0][0], LDP);
    __syncthreads();
    pack_rm(&Cf[0][0], LDP, Apk);          // A bf16 (symmetric: row pack serves B too)
    __syncthreads();
    // Y1 = A X : B(k,n) = X[k][n] -> B^T packed == XpkT
    block_gemm64_pk(Apk, XpkT, &Cf[0][0], LDP);
    __syncthreads();
    pack_tr(&Cf[0][0], LDP, XpkR);         // Y1^T packed (XpkR slot is dead this iter)
    __syncthreads();
    // Y2 = A Y1, fused: X = a*X + b*Y1 + c*Y2
    block_gemm64_pk_polar(Apk, XpkR, &Cf[0][0], LDP, &Xs[0][0], LDP);
    __syncthreads();
    if (it < NSTEP - 1) {
      pack_rm(&Xs[0][0], LDP, XpkR);
      pack_tr(&Xs[0][0], LDP, XpkT);
      __syncthreads();
    }
  }
  for (int i = tid; i < 4096; i += 256) mat[i] = Xs[i >> 6][i & 63];
}

// alpha-gated M scan (registers) + fused matvec y_t = M_t q_t
__global__ __launch_bounds__(256) void k_apply(
    const float* __restrict__ qb, float* __restrict__ Mst,
    const float* __restrict__ cS, const float* __restrict__ gal,
    float* __restrict__ yb, int chunk) {
  __shared__ float Qs[64][LDP];
  __shared__ float al[CSZ];
  __shared__ float yred[64][4];
  const int bh = blockIdx.x, b = bh >> 3, h = bh & 7;
  const int t0 = chunk * CSZ, tid = threadIdx.x;
#ifdef HAVE_TDM
  if (tid == 0) {
    tdm_load_tile64(qb + ((size_t)(b * TT + t0) * HN + h) * DD,
                    LDS_BYTE_OFF(&Qs[0][0]), HN * DD);
  }
  __builtin_amdgcn_s_wait_tensorcnt(0);
#else
  for (int i = tid; i < 4096; i += 256) {
    const int t = i >> 6, d = i & 63;
    Qs[t][d] = qb[((size_t)(b * TT + t0 + t) * HN + h) * DD + d];
  }
#endif
  if (tid < CSZ) al[tid] = gal[(size_t)(b * TT + t0 + tid) * HN + h];
  float M[16];
  const size_t mb = (size_t)bh * 4096 + (size_t)tid * 16;
#pragma unroll
  for (int j = 0; j < 16; ++j) M[j] = Mst[mb + j];
  __syncthreads();
  const int v = tid >> 2, c4 = tid & 3, seg = c4 << 4;
  const float* Zb = cS + (size_t)bh * CSZ * 4096 + (size_t)tid * 16;
  for (int t = 0; t < CSZ; ++t) {
    const float a = al[t];
    const float* Z = Zb + (size_t)t * 4096;
    if (t + 1 < CSZ) __builtin_prefetch(Zb + (size_t)(t + 1) * 4096, 0, 0);
    float part = 0.f;
#pragma unroll
    for (int j = 0; j < 16; ++j) {
      const float m = a * M[j] + Z[j];
      M[j] = m;
      part += m * Qs[t][seg + j];
    }
    yred[v][c4] = part;
    __syncthreads();
    if (c4 == 0) {
      const float yv = yred[v][0] + yred[v][1] + yred[v][2] + yred[v][3];
      yb[((size_t)(b * TT + t0 + t) * HN + h) * DD + v] = yv;
    }
    __syncthreads();
  }
#pragma unroll
  for (int j = 0; j < 16; ++j) Mst[mb + j] = M[j];
}

// rms_norm over D, * (1+ln_gamma), * retrieve gate (in place)
__global__ __launch_bounds__(64) void k_norm(float* __restrict__ y,
                                             const float* __restrict__ lng,
                                             const float* __restrict__ rg) {
  __shared__ float sq[64];
  const int g = blockIdx.x, d = threadIdx.x;
  const int h = g & 7;
  const float v = y[(size_t)g * 64 + d];
  sq[d] = v * v;
  __syncthreads();
  float s = 0.f;
#pragma unroll
  for (int i = 0; i < 64; ++i) s += sq[i];
  const float r = rsqrtf(s * (1.0f / 64.f) + 1e-6f);
  y[(size_t)g * 64 + d] = v * r * (1.f + lng[h * 64 + d]) * rg[g];
}

// ---------------- launcher ----------------
extern "C" void kernel_launch(void* const* d_in, const int* in_sizes, int n_in,
                              void* d_out, int out_size, void* d_ws, size_t ws_size,
                              hipStream_t stream) {
  (void)in_sizes; (void)n_in; (void)out_size; (void)ws_size;
  const float* x     = (const float*)d_in[0];
  const float* Wq    = (const float*)d_in[1];
  const float* Wk    = (const float*)d_in[2];
  const float* Wv    = (const float*)d_in[3];
  const float* Wproj = (const float*)d_in[4];
  const float* cqw   = (const float*)d_in[5];
  const float* cqb   = (const float*)d_in[6];
  const float* ckw   = (const float*)d_in[7];
  const float* ckb   = (const float*)d_in[8];
  const float* cvw   = (const float*)d_in[9];
  const float* cvb   = (const float*)d_in[10];
  const float* gaw   = (const float*)d_in[11];
  const float* gab   = (const float*)d_in[12];
  const float* gew   = (const float*)d_in[13];
  const float* geb   = (const float*)d_in[14];
  const float* gtw   = (const float*)d_in[15];
  const float* gtb   = (const float*)d_in[16];
  const float* ggw   = (const float*)d_in[17];
  const float* ggb   = (const float*)d_in[18];
  const float* poly  = (const float*)d_in[19];
  const float* lng   = (const float*)d_in[20];
  const float* W1    = (const float*)d_in[21];
  const float* rgw   = (const float*)d_in[22];

  float* ws = (float*)d_ws;
  const size_t NBTHD = (size_t)BB * TT * HN * DD;      // 1048576
  float* qbuf   = ws;
  float* kbuf   = qbuf + NBTHD;
  float* vbuf   = kbuf + NBTHD;
  float* ybuf   = vbuf + NBTHD;
  float* galpha = ybuf + NBTHD;
  float* geta   = galpha + BB * TT * HN;
  float* gtheta = geta   + BB * TT * HN;
  float* ggamma = gtheta + BB * TT * HN;
  float* grg    = ggamma + BB * TT * HN;
  float* Mst    = grg    + BB * TT * HN;               // B*H*64*64
  float* Sst    = Mst + BB * HN * DD * DD;
  float* chunkS = Sst + BB * HN * DD * DD;             // B*H*CS*64*64 = 16MB
  // pre-conv q/k/v buffers overlay chunkS (dead before chunk loop starts)
  float* preq = chunkS;
  float* prek = preq + NBTHD;
  float* prev = prek + NBTHD;

  // carries
  k_init<<<256, 256, 0, stream>>>(W1, Mst, Sst);

  // q/k/v projections (WMMA GEMM)
  dim3 pg(BB * TT / 64, CC / 64);
  k_proj<<<pg, 256, 0, stream>>>(x, Wq, preq, CC);
  k_proj<<<pg, 256, 0, stream>>>(x, Wk, prek, CC);
  k_proj<<<pg, 256, 0, stream>>>(x, Wv, prev, CC);

  // causal depthwise conv + phi
  const int convBlocks = BB * TT * CC / 256;
  k_conv<<<convBlocks, 256, 0, stream>>>(preq, cqw, cqb, poly, qbuf, 1);
  k_conv<<<convBlocks, 256, 0, stream>>>(prek, ckw, ckb, poly, kbuf, 1);
  k_conv<<<convBlocks, 256, 0, stream>>>(prev, cvw, cvb, poly, vbuf, 0);

  // gates
  k_gates<<<BB * TT, 64, 0, stream>>>(x, gaw, gab, gew, geb, gtw, gtb,
                                      ggw, ggb, rgw,
                                      galpha, geta, gtheta, ggamma, grg);

  // chunk recurrence
  for (int c = 0; c < NCHUNK; ++c) {
    k_err_scan<<<BB * HN, 256, 0, stream>>>(kbuf, vbuf, Mst, Sst, chunkS,
                                            gtheta, geta, ggamma, c);
    k_ns<<<BB * HN * CSZ, 256, 0, stream>>>(chunkS);
    k_apply<<<BB * HN, 256, 0, stream>>>(qbuf, Mst, chunkS, galpha, ybuf, c);
  }

  // norm + retrieve gate, then output projection (WMMA GEMM) into d_out
  k_norm<<<BB * TT * HN, 64, 0, stream>>>(ybuf, lng, grg);
  k_proj<<<pg, 256, 0, stream>>>(ybuf, Wproj, (float*)d_out, CC);
}
